// Transformer_10917806866949
// MI455X (gfx1250) — matrix operand order repaired and checked by reading, not compiled, not run on hardware
//
#include <hip/hip_runtime.h>

// ---------------- constants ----------------
constexpr int kL    = 4;
constexpr int kD    = 512;
constexpr int kH    = 8;
constexpr int kB    = 128;
constexpr int kNT   = 512;
constexpr int kS    = kNT + 1;          // 513
constexpr int kDHID = 682;              // int(512*4/3)
constexpr int kKP1  = 704;              // kDHID padded to 22*32 for aligned WMMA K-loop
constexpr long long kMS = (long long)kB * kS;   // 65664 rows

typedef __attribute__((ext_vector_type(16))) __bf16 v16bf;
typedef __attribute__((ext_vector_type(8)))  float  v8f;
typedef __attribute__((ext_vector_type(4)))  unsigned int v4u;
typedef __attribute__((ext_vector_type(8)))  int v8i;
typedef __attribute__((ext_vector_type(4)))  int v4i;

union Frag { v16bf v; uint4 q[2]; unsigned short u[16]; };
static_assert(sizeof(Frag) == 32, "frag size");

__device__ __forceinline__ unsigned short f2bf(float f) {
  union { float f; unsigned u; } c; c.f = f;
  unsigned u = c.u;
  unsigned r = u + 0x7FFFu + ((u >> 16) & 1u);   // RNE
  return (unsigned short)(r >> 16);
}

__device__ __forceinline__ v8f wmma_bf16(const v16bf& a, const v16bf& b, const v8f& c) {
  return __builtin_amdgcn_wmma_f32_16x16x32_bf16(false, a, false, b, (short)0, c, false, false);
}

// ---- Tensor Data Mover: 2D tile (tile1 rows x tile0 elems, 2-byte elements)
// from global (row stride = strideElems elements) into LDS at byte offset ldsOff.
// D# layout per CDNA5 ISA ch.8 (group0: count/lds/global/type=2; group1: dims/strides).
// 6-arg builtin form: (v4u g0, v8i g1, v4i g2, v4i g3, v8i g4, i32 cpol).
__device__ __forceinline__ void tdm_load_2d(unsigned ldsOff, const void* gptr,
                                            int tile0, int tile1,
                                            int tdim0, int tdim1,
                                            long long strideElems) {
  unsigned long long ga = (unsigned long long)(size_t)gptr;
  v4u g0;
  g0[0] = 1u;                                   // count=1, user descriptor
  g0[1] = ldsOff;                               // lds_addr (bytes)
  g0[2] = (unsigned)ga;                         // global_addr[31:0]
  g0[3] = (unsigned)(ga >> 32) | (2u << 30);    // global_addr[56:32] | type=2
  v8i g1;
  g1[0] = (int)(1u << 16);                      // data_size=1 (2 bytes); no pad/iterate/multicast
  g1[1] = (int)(((unsigned)tdim0 & 0xFFFFu) << 16);                 // tensor_dim0[15:0]
  g1[2] = (int)((((unsigned)tdim0 >> 16) & 0xFFFFu) |
                (((unsigned)tdim1 & 0xFFFFu) << 16));               // dim0[31:16] | dim1[15:0]
  g1[3] = (int)((((unsigned)tdim1 >> 16) & 0xFFFFu) |
                ((unsigned)tile0 << 16));                           // dim1[31:16] | tile_dim0
  g1[4] = (int)((unsigned)tile1 & 0xFFFFu);                         // tile_dim1 (tile_dim2=0)
  g1[5] = (int)(unsigned)(strideElems & 0xFFFFFFFFll);              // dim0_stride[31:0]
  g1[6] = (int)(unsigned)((strideElems >> 32) & 0xFFFFll);          // dim0_stride[47:32]
  g1[7] = 0;
  v4i z4; z4[0] = 0; z4[1] = 0; z4[2] = 0; z4[3] = 0;
  v8i z8; z8[0] = 0; z8[1] = 0; z8[2] = 0; z8[3] = 0;
  z8[4] = 0; z8[5] = 0; z8[6] = 0; z8[7] = 0;
  __builtin_amdgcn_tensor_load_to_lds(g0, g1, z4, z4, z8, 0);
}

// ---------------- elementwise helpers ----------------
__global__ __launch_bounds__(256) void cvt_kernel(const float* __restrict__ src,
                                                  unsigned short* __restrict__ dst,
                                                  long long n) {
  long long i = (long long)blockIdx.x * 256 + threadIdx.x;
  if (i < n) dst[i] = f2bf(src[i]);
}

// convert (rows,K) f32 -> (rows,Kp) bf16, zero padding cols K..Kp-1
__global__ __launch_bounds__(256) void cvt_pad_kernel(const float* __restrict__ src,
                                                      unsigned short* __restrict__ dst,
                                                      int rows, int K, int Kp) {
  long long i = (long long)blockIdx.x * 256 + threadIdx.x;
  long long n = (long long)rows * Kp;
  if (i >= n) return;
  int c = (int)(i % Kp);
  long long r = i / Kp;
  dst[i] = (c < K) ? f2bf(src[r * K + c]) : (unsigned short)0;
}

// x = concat(cls, x_num) -> (B, S, D) f32
__global__ __launch_bounds__(256) void concat_kernel(const float* __restrict__ x_num,
                                                     const float* __restrict__ cls,
                                                     float* __restrict__ x) {
  long long idx = (long long)blockIdx.x * 256 + threadIdx.x;
  long long n = kMS * kD;
  if (idx >= n) return;
  int c = (int)(idx % kD);
  long long r = idx / kD;
  int srow = (int)(r % kS);
  long long b = r / kS;
  x[idx] = (srow == 0) ? cls[c] : x_num[((b * kNT) + srow - 1) * kD + c];
}

// one wave per row LayerNorm, bf16 output (feeds WMMA GEMMs)
__global__ __launch_bounds__(256) void ln_kernel(const float* __restrict__ in,
                                                 const float* __restrict__ g,
                                                 const float* __restrict__ bt,
                                                 unsigned short* __restrict__ out,
                                                 int M) {
  int wv = threadIdx.x >> 5, lane = threadIdx.x & 31;
  int row = blockIdx.x * 8 + wv;
  if (row >= M) return;
  const float* p = in + (long long)row * kD;
  float xv[16], s = 0.f, s2 = 0.f;
#pragma unroll
  for (int t = 0; t < 16; ++t) { xv[t] = p[t * 32 + lane]; s += xv[t]; s2 += xv[t] * xv[t]; }
#pragma unroll
  for (int off = 16; off >= 1; off >>= 1) { s += __shfl_xor(s, off, 32); s2 += __shfl_xor(s2, off, 32); }
  float mean = s * (1.f / kD);
  float var  = s2 * (1.f / kD) - mean * mean;
  float rst  = rsqrtf(var + 1e-5f);
#pragma unroll
  for (int t = 0; t < 16; ++t) {
    int c = t * 32 + lane;
    out[(long long)row * kD + c] = f2bf((xv[t] - mean) * rst * g[c] + bt[c]);
  }
}

// final: LN + relu + dot(head_w) + head_b  -> out[row]
__global__ __launch_bounds__(32) void head_kernel(const float* __restrict__ xB,
                                                  const float* __restrict__ g,
                                                  const float* __restrict__ bt,
                                                  const float* __restrict__ hw,
                                                  const float* __restrict__ hb,
                                                  float* __restrict__ out, int M) {
  int row = blockIdx.x, lane = threadIdx.x;
  if (row >= M) return;
  const float* p = xB + (long long)row * kD;
  float xv[16], s = 0.f, s2 = 0.f;
#pragma unroll
  for (int t = 0; t < 16; ++t) { xv[t] = p[t * 32 + lane]; s += xv[t]; s2 += xv[t] * xv[t]; }
#pragma unroll
  for (int off = 16; off >= 1; off >>= 1) { s += __shfl_xor(s, off, 32); s2 += __shfl_xor(s2, off, 32); }
  float mean = s * (1.f / kD);
  float rst  = rsqrtf(s2 * (1.f / kD) - mean * mean + 1e-5f);
  float dot = 0.f;
#pragma unroll
  for (int t = 0; t < 16; ++t) {
    int c = t * 32 + lane;
    float v = (xv[t] - mean) * rst * g[c] + bt[c];
    dot += fmaxf(v, 0.f) * hw[c];
  }
#pragma unroll
  for (int off = 16; off >= 1; off >>= 1) dot += __shfl_xor(dot, off, 32);
  if (lane == 0) out[row] = dot + hb[0];
}

// ---------------- WMMA GEMM: C(M,N) = A(M,K)bf16 @ W(N,K)bf16^T + bias ----------------
// 128x128 block tile, 8 waves each owning 32x64 (2x4 WMMA tiles).
// K staged in 32-wide chunks by the Tensor Data Mover, double-buffered in LDS:
// wave 0 issues tensor_load_to_lds for chunk k+1 while all waves compute chunk k,
// then drains TENSORcnt and the block barrier publishes the tile.
// MODE 0: outF = acc+bias                    (f32)
// MODE 1: outF = acc+bias + res              (f32, residual add)
// MODE 2: outB = bf16(acc+bias)              (bf16 out for next GEMM)
// MODE 3: outB = bf16(aux * relu(acc+bias))  (fused ReGLU combine)
template <int MODE>
__global__ __launch_bounds__(256) void gemm_kernel(
    const unsigned short* __restrict__ A, long long strideA,
    const unsigned short* __restrict__ W, int ldW,
    const float* __restrict__ bias,
    float* __restrict__ outF, unsigned short* __restrict__ outB, int strideOut,
    const float* __restrict__ res, long long strideRes,
    const float* __restrict__ aux, int strideAux,
    int M, int N, int K) {
  __shared__ __align__(16) unsigned short sA[2][128 * 32];
  __shared__ __align__(16) unsigned short sB[2][128 * 32];

  const int tid = threadIdx.x;
  const int m0 = blockIdx.y * 128;
  const int n0 = blockIdx.x * 128;
  const int lane = tid & 31, wv = tid >> 5;
  const int lr = lane & 15, lh = lane >> 4;
  const int wm = wv >> 1, wn = wv & 1;

  v8f acc[2][4];
#pragma unroll
  for (int i = 0; i < 2; ++i)
#pragma unroll
    for (int j = 0; j < 4; ++j)
#pragma unroll
      for (int e = 0; e < 8; ++e) acc[i][j][e] = 0.f;

  const unsigned ldsA[2] = { (unsigned)(size_t)(void*)&sA[0][0],
                             (unsigned)(size_t)(void*)&sA[1][0] };
  const unsigned ldsB[2] = { (unsigned)(size_t)(void*)&sB[0][0],
                             (unsigned)(size_t)(void*)&sB[1][0] };
  const int nk = K / 32;
  const int remM = M - m0;
  const int remN = N - n0;

  if (wv == 0) {   // prologue: DMA chunk 0
    tdm_load_2d(ldsA[0], A + (long long)m0 * strideA, 32, 128, 32, remM, strideA);
    tdm_load_2d(ldsB[0], W + (long long)n0 * ldW,     32, 128, 32, remN, (long long)ldW);
    __builtin_amdgcn_s_wait_tensorcnt(0);
  }
  __syncthreads();

  for (int kc = 0; kc < nk; ++kc) {
    const int cur = kc & 1, nxt = cur ^ 1;
    if (kc + 1 < nk && wv == 0) {   // DMA next chunk while computing this one
      tdm_load_2d(ldsA[nxt], A + (long long)m0 * strideA + (kc + 1) * 32, 32, 128, 32, remM, strideA);
      tdm_load_2d(ldsB[nxt], W + (long long)n0 * ldW + (kc + 1) * 32,     32, 128, 32, remN, (long long)ldW);
    }

    const unsigned short* pa = &sA[cur][0];
    const unsigned short* pb = &sB[cur][0];
    Frag fa[2], fb[4];
#pragma unroll
    for (int i = 0; i < 2; ++i) {   // A 16x32: lane=row, K halves split by lane-half
      int base = (wm * 32 + i * 16 + lr) * 32 + lh * 8;
      fa[i].q[0] = *(const uint4*)&pa[base];
      fa[i].q[1] = *(const uint4*)&pa[base + 16];
    }
#pragma unroll
    for (int j = 0; j < 4; ++j) {   // B 32x16: lane=col, K halves split by lane-half
      int base = (wn * 64 + j * 16 + lr) * 32 + lh * 8;
      fb[j].q[0] = *(const uint4*)&pb[base];
      fb[j].q[1] = *(const uint4*)&pb[base + 16];
    }
#pragma unroll
    for (int i = 0; i < 2; ++i)
#pragma unroll
      for (int j = 0; j < 4; ++j)
        acc[i][j] = wmma_bf16(fa[i].v, fb[j].v, acc[i][j]);

    if (wv == 0) __builtin_amdgcn_s_wait_tensorcnt(0);
    __syncthreads();
  }

#pragma unroll
  for (int i = 0; i < 2; ++i)
#pragma unroll
    for (int j = 0; j < 4; ++j) {
      int col = n0 + wn * 64 + j * 16 + lr;
#pragma unroll
      for (int v = 0; v < 8; ++v) {   // D tile: row = v + 8*lh
        int row = m0 + wm * 32 + i * 16 + v + 8 * lh;
        if (row < M && col < N) {
          float val = acc[i][j][v] + bias[col];
          if (MODE == 0) {
            outF[(long long)row * strideOut + col] = val;
          } else if (MODE == 1) {
            outF[(long long)row * strideOut + col] = val + res[(long long)row * strideRes + col];
          } else if (MODE == 2) {
            outB[(long long)row * strideOut + col] = f2bf(val);
          } else {
            float a = aux[(long long)row * strideAux + col];
            outB[(long long)row * strideOut + col] = f2bf(a * fmaxf(val, 0.f));
          }
        }
      }
    }
}

// ---------------- flash-style WMMA attention ----------------
// q: (B*Sq, D) bf16 ; k,v: (B*S, D) bf16 ; out: (B*Sq, D) bf16
// grid(B*H, ceil(Sq/64)), 128 threads (4 waves x 16 queries)
__global__ __launch_bounds__(128) void attn_kernel(
    const unsigned short* __restrict__ qb,
    const unsigned short* __restrict__ kb,
    const unsigned short* __restrict__ vbuf,
    unsigned short* __restrict__ ob,
    int Sq) {
  __shared__ __align__(16) unsigned short sK[32 * 64];    // [key][c]
  __shared__ __align__(16) unsigned short sVT[64 * 32];   // [c][key]
  __shared__ __align__(16) unsigned short sP[4][16 * 32]; // per-wave P tile

  const int tid = threadIdx.x;
  const int lane = tid & 31, wv = tid >> 5;
  const int lr = lane & 15, lh = lane >> 4;
  const int b = blockIdx.x / kH, h = blockIdx.x % kH;
  const int qbase = blockIdx.y * 64 + wv * 16;

  // Q fragments (K-dim = dh = 64 -> two 16x32 A-frags), loaded once from global
  Frag qf[2];
  {
    int qi = qbase + lr; if (qi >= Sq) qi = Sq - 1;
    const unsigned short* qp = qb + ((long long)b * Sq + qi) * kD + h * 64;
#pragma unroll
    for (int dc = 0; dc < 2; ++dc) {
      qf[dc].q[0] = *(const uint4*)(qp + dc * 32 + lh * 8);
      qf[dc].q[1] = *(const uint4*)(qp + dc * 32 + lh * 8 + 16);
    }
  }

  float mrow[8], lrow[8];
  v8f o[4];
#pragma unroll
  for (int v = 0; v < 8; ++v) { mrow[v] = -1e30f; lrow[v] = 0.f; }
#pragma unroll
  for (int jn = 0; jn < 4; ++jn)
#pragma unroll
    for (int e = 0; e < 8; ++e) o[jn][e] = 0.f;

  const int nkc = (kS + 31) / 32;   // 17 chunks of 32 keys
  for (int kc = 0; kc < nkc; ++kc) {
    __syncthreads();
    {   // stage K chunk (32x64) and V chunk transposed (64x32)
      int krow = tid >> 2;            // 0..31
      int cs   = (tid & 3) * 16;      // 0,16,32,48
      int key  = kc * 32 + krow;
      uint4 d0 = {0, 0, 0, 0}, d1 = {0, 0, 0, 0};
      if (key < kS) {
        const uint4* p = (const uint4*)(kb + ((long long)b * kS + key) * kD + h * 64 + cs);
        d0 = p[0]; d1 = p[1];
      }
      *(uint4*)&sK[krow * 64 + cs]     = d0;
      *(uint4*)&sK[krow * 64 + cs + 8] = d1;

      union { uint4 q[2]; unsigned short u[16]; } vd;
      vd.q[0] = make_uint4(0, 0, 0, 0); vd.q[1] = make_uint4(0, 0, 0, 0);
      if (key < kS) {
        const uint4* p = (const uint4*)(vbuf + ((long long)b * kS + key) * kD + h * 64 + cs);
        vd.q[0] = p[0]; vd.q[1] = p[1];
      }
#pragma unroll
      for (int e = 0; e < 16; ++e) sVT[(cs + e) * 32 + krow] = vd.u[e];
    }
    __syncthreads();

    // scores S = Q @ K^T * scale : two 16x16 tiles over this 32-key chunk
    v8f s0, s1;
#pragma unroll
    for (int e = 0; e < 8; ++e) { s0[e] = 0.f; s1[e] = 0.f; }
#pragma unroll
    for (int dc = 0; dc < 2; ++dc) {
      Frag kf0, kf1;
      int b0 = (lr) * 64 + dc * 32 + lh * 8;         // key col 0..15
      kf0.q[0] = *(const uint4*)&sK[b0];
      kf0.q[1] = *(const uint4*)&sK[b0 + 16];
      int b1 = (16 + lr) * 64 + dc * 32 + lh * 8;    // key col 16..31
      kf1.q[0] = *(const uint4*)&sK[b1];
      kf1.q[1] = *(const uint4*)&sK[b1 + 16];
      s0 = wmma_bf16(qf[dc].v, kf0.v, s0);
      s1 = wmma_bf16(qf[dc].v, kf1.v, s1);
    }
    const bool ok0 = (kc * 32 + lr) < kS;
    const bool ok1 = (kc * 32 + 16 + lr) < kS;
#pragma unroll
    for (int v = 0; v < 8; ++v) {
      s0[v] = ok0 ? s0[v] * 0.125f : -1e30f;
      s1[v] = ok1 ? s1[v] * 0.125f : -1e30f;
    }

    // online softmax (rows live in vgpr index + lane-half; cols across 16 lanes)
#pragma unroll
    for (int v = 0; v < 8; ++v) {
      float cm = fmaxf(s0[v], s1[v]);
#pragma unroll
      for (int off = 8; off >= 1; off >>= 1) cm = fmaxf(cm, __shfl_xor(cm, off, 32));
      float mn   = fmaxf(mrow[v], cm);
      float corr = __expf(mrow[v] - mn);
      float p0   = __expf(s0[v] - mn);
      float p1   = __expf(s1[v] - mn);
      float rs   = p0 + p1;
#pragma unroll
      for (int off = 8; off >= 1; off >>= 1) rs += __shfl_xor(rs, off, 32);
      lrow[v] = lrow[v] * corr + rs;
      mrow[v] = mn;
#pragma unroll
      for (int jn = 0; jn < 4; ++jn) o[jn][v] *= corr;
      int prow = v + 8 * lh;
      sP[wv][prow * 32 + lr]      = f2bf(p0);
      sP[wv][prow * 32 + 16 + lr] = f2bf(p1);
    }
    asm volatile("s_wait_dscnt 0" ::: "memory");   // wave-local P turnaround

    // P (16x32) A-frag, then O += P @ V over 4 dh-tiles
    Frag pf;
    {
      int pb = lr * 32 + lh * 8;
      pf.q[0] = *(const uint4*)&sP[wv][pb];
      pf.q[1] = *(const uint4*)&sP[wv][pb + 16];
    }
#pragma unroll
    for (int jn = 0; jn < 4; ++jn) {
      Frag vf;
      int vb0 = (jn * 16 + lr) * 32 + lh * 8;    // sVT[c][key]
      vf.q[0] = *(const uint4*)&sVT[vb0];
      vf.q[1] = *(const uint4*)&sVT[vb0 + 16];
      o[jn] = wmma_bf16(pf.v, vf.v, o[jn]);
    }
  }

  // normalize + store bf16 into (B*Sq, D) at head offset
#pragma unroll
  for (int jn = 0; jn < 4; ++jn) {
    int col = h * 64 + jn * 16 + lr;
#pragma unroll
    for (int v = 0; v < 8; ++v) {
      int qi = qbase + v + 8 * lh;
      if (qi < Sq)
        ob[((long long)b * Sq + qi) * kD + col] = f2bf(o[jn][v] / lrow[v]);
    }
  }
}

// ---------------- host orchestration ----------------
extern "C" void kernel_launch(void* const* d_in, const int* in_sizes, int n_in,
                              void* d_out, int out_size, void* d_ws, size_t ws_size,
                              hipStream_t stream) {
  const float* x_num = (const float*)d_in[0];
  const float* cls_w = (const float*)d_in[1];
  const float* Wq = (const float*)d_in[2];  const float* bq = (const float*)d_in[3];
  const float* Wk = (const float*)d_in[4];  const float* bk = (const float*)d_in[5];
  const float* Wv = (const float*)d_in[6];  const float* bv = (const float*)d_in[7];
  const float* Wo = (const float*)d_in[8];  const float* bo = (const float*)d_in[9];
  const float* l0_w = (const float*)d_in[10]; const float* l0_b = (const float*)d_in[11];
  const float* l1_w = (const float*)d_in[12]; const float* l1_b = (const float*)d_in[13];
  const float* n0_g = (const float*)d_in[14]; const float* n0_b = (const float*)d_in[15];
  const float* n1_g = (const float*)d_in[16]; const float* n1_b = (const float*)d_in[17];
  const float* ln_g = (const float*)d_in[18]; const float* ln_b = (const float*)d_in[19];
  const float* head_w = (const float*)d_in[20]; const float* head_b = (const float*)d_in[21];

  char* wsp = (char*)d_ws;
  size_t off = 0;
  auto carve = [&](size_t bytes) -> char* {
    char* p = wsp + off;
    off += (bytes + 255) & ~(size_t)255;
    return p;
  };
  const size_t DD = (size_t)kD * kD;
  unsigned short* wqb  = (unsigned short*)carve(kL * DD * 2);
  unsigned short* wkb  = (unsigned short*)carve(kL * DD * 2);
  unsigned short* wvb  = (unsigned short*)carve(kL * DD * 2);
  unsigned short* wob  = (unsigned short*)carve(kL * DD * 2);
  unsigned short* l0bw = (unsigned short*)carve((size_t)kL * 2 * kDHID * kD * 2);
  unsigned short* l1bw = (unsigned short*)carve((size_t)kL * kD * kKP1 * 2);
  float*          xfull = (float*)carve((size_t)kMS * kD * 4);
  float*          xBuf  = (float*)carve((size_t)kB * kD * 4);
  unsigned short* xn  = (unsigned short*)carve((size_t)kMS * kD * 2);
  unsigned short* qbb = (unsigned short*)carve((size_t)kMS * kD * 2);
  unsigned short* kbb = (unsigned short*)carve((size_t)kMS * kD * 2);
  unsigned short* vbb = (unsigned short*)carve((size_t)kMS * kD * 2);
  unsigned short* xat = (unsigned short*)carve((size_t)kMS * kD * 2);
  float*          ha  = (float*)carve((size_t)kMS * kDHID * 4);
  unsigned short* hg  = (unsigned short*)carve((size_t)kMS * kKP1 * 2);

  auto cvtN = [&](const float* s, unsigned short* d, long long n) {
    cvt_kernel<<<(unsigned)((n + 255) / 256), 256, 0, stream>>>(s, d, n);
  };

  // weight conversion (bf16, l1 K-padded to 704)
  cvtN(Wq, wqb, (long long)kL * DD);
  cvtN(Wk, wkb, (long long)kL * DD);
  cvtN(Wv, wvb, (long long)kL * DD);
  cvtN(Wo, wob, (long long)kL * DD);
  cvtN(l0_w, l0bw, (long long)kL * 2 * kDHID * kD);
  for (int i = 0; i < kL; ++i) {
    long long n = (long long)kD * kKP1;
    cvt_pad_kernel<<<(unsigned)((n + 255) / 256), 256, 0, stream>>>(
        l1_w + (size_t)i * kD * kDHID, l1bw + (size_t)i * kD * kKP1, kD, kDHID, kKP1);
  }

  // build x = concat(cls, x_num)
  {
    long long n = kMS * kD;
    concat_kernel<<<(unsigned)((n + 255) / 256), 256, 0, stream>>>(x_num, cls_w, xfull);
  }

  for (int i = 0; i < kL; ++i) {
    const bool last = (i == kL - 1);
    const int Sq = last ? 1 : kS;
    const int Mq = kB * Sq;
    const int MSi = (int)kMS;

    // pre-norm (layer 0: raw copy to bf16)
    if (i == 0) cvtN(xfull, xn, kMS * kD);
    else ln_kernel<<<(MSi + 7) / 8, 256, 0, stream>>>(xfull, n0_g + (size_t)(i - 1) * kD,
                                                      n0_b + (size_t)(i - 1) * kD, xn, MSi);

    // Q/K/V projections -> bf16
    long long qStrideA = last ? (long long)kS * kD : kD;
    gemm_kernel<2><<<dim3((kD + 127) / 128, (Mq + 127) / 128), 256, 0, stream>>>(
        xn, qStrideA, wqb + (size_t)i * DD, kD, bq + (size_t)i * kD,
        nullptr, qbb, kD, nullptr, 0, nullptr, 0, Mq, kD, kD);
    gemm_kernel<2><<<dim3((kD + 127) / 128, (MSi + 127) / 128), 256, 0, stream>>>(
        xn, (long long)kD, wkb + (size_t)i * DD, kD, bk + (size_t)i * kD,
        nullptr, kbb, kD, nullptr, 0, nullptr, 0, MSi, kD, kD);
    gemm_kernel<2><<<dim3((kD + 127) / 128, (MSi + 127) / 128), 256, 0, stream>>>(
        xn, (long long)kD, wvb + (size_t)i * DD, kD, bv + (size_t)i * kD,
        nullptr, vbb, kD, nullptr, 0, nullptr, 0, MSi, kD, kD);

    // attention
    attn_kernel<<<dim3(kB * kH, (Sq + 63) / 64), 128, 0, stream>>>(qbb, kbb, vbb, xat, Sq);

    // output projection + residual  (last layer shrinks to cls row)
    float* outx = last ? xBuf : xfull;
    long long resStride = last ? (long long)kS * kD : kD;
    gemm_kernel<1><<<dim3((kD + 127) / 128, (Mq + 127) / 128), 256, 0, stream>>>(
        xat, (long long)kD, wob + (size_t)i * DD, kD, bo + (size_t)i * kD,
        outx, nullptr, kD, xfull, resStride, nullptr, 0, Mq, kD, kD);

    // FFN with ReGLU
    float* xcur = outx;
    ln_kernel<<<(Mq + 7) / 8, 256, 0, stream>>>(xcur, n1_g + (size_t)i * kD,
                                                n1_b + (size_t)i * kD, xn, Mq);
    const unsigned short* l0a = l0bw + (size_t)i * 2 * kDHID * kD;
    const unsigned short* l0b2 = l0a + (size_t)kDHID * kD;
    gemm_kernel<0><<<dim3((kDHID + 127) / 128, (Mq + 127) / 128), 256, 0, stream>>>(
        xn, (long long)kD, l0a, kD, l0_b + (size_t)i * 2 * kDHID,
        ha, nullptr, kDHID, nullptr, 0, nullptr, 0, Mq, kDHID, kD);
    gemm_kernel<3><<<dim3((kDHID + 127) / 128, (Mq + 127) / 128), 256, 0, stream>>>(
        xn, (long long)kD, l0b2, kD, l0_b + (size_t)i * 2 * kDHID + kDHID,
        nullptr, hg, kKP1, nullptr, 0, ha, kDHID, Mq, kDHID, kD);
    gemm_kernel<1><<<dim3((kD + 127) / 128, (Mq + 127) / 128), 256, 0, stream>>>(
        hg, (long long)kKP1, l1bw + (size_t)i * kD * kKP1, kKP1, l1_b + (size_t)i * kD,
        xcur, nullptr, kD, xcur, (long long)kD, nullptr, 0, Mq, kD, kKP1);
  }

  head_kernel<<<kB, 32, 0, stream>>>(xBuf, ln_g, ln_b, head_w, head_b, (float*)d_out, kB);
}
